// Transformer_73443940762166
// MI455X (gfx1250) — matrix operand order repaired and checked by reading, not compiled, not run on hardware
//
#include <hip/hip_runtime.h>
#include <hip/hip_bf16.h>

// ---------------------------------------------------------------------------
// Types
// ---------------------------------------------------------------------------
typedef __attribute__((ext_vector_type(16))) __bf16 v16bf;
typedef __attribute__((ext_vector_type(8)))  __bf16 bf16x8;
typedef __attribute__((ext_vector_type(8)))  float  v8f;
typedef __attribute__((ext_vector_type(4)))  float  f32x4;
typedef __attribute__((ext_vector_type(4)))  unsigned u32x4;

// ---------------------------------------------------------------------------
// Problem dims (fixed by the reference)
// ---------------------------------------------------------------------------
static constexpr int TB  = 4;      // batch
static constexpr int TS  = 1024;   // seq len
static constexpr int TD  = 512;    // d_model
static constexpr int THd = 8;      // heads
static constexpr int TL  = 6;      // layers
static constexpr int TF  = 2048;   // ffn hidden
static constexpr int TV  = 25000;  // vocab
static constexpr int TDK = 64;     // head dim
static constexpr int PAD_TOK = 1;

// GEMM tiling: 128x128 block tile, 256 threads = 8 waves, wave = 64x32
#define BM 128
#define BN 128
#define BKt 32
#define BKP 40   // padded K stride (halves): 80 bytes/row, keeps 16B alignment

// ---------------------------------------------------------------------------
// fp32 -> bf16 packing.
// Fast path: round-to-nearest (+0x8000) then a single v_perm_b32 packs the two
// high halves of (lo, hi) into one dword. Fallback: manual RNE.
// ---------------------------------------------------------------------------
__device__ __forceinline__ unsigned short f2bf_u(float f) {
  unsigned u = __builtin_bit_cast(unsigned, f);
  u += 0x7FFFu + ((u >> 16) & 1u);
  return (unsigned short)(u >> 16);
}

__device__ __forceinline__ unsigned pkbf2(float lo, float hi) {
#if __has_builtin(__builtin_amdgcn_perm)
  unsigned ul = __builtin_bit_cast(unsigned, lo) + 0x8000u;
  unsigned uh = __builtin_bit_cast(unsigned, hi) + 0x8000u;
  // result bytes: [0]=ul.b2 [1]=ul.b3 [2]=uh.b2 [3]=uh.b3
  return __builtin_amdgcn_perm(uh, ul, 0x07060302u);
#else
  return (unsigned)f2bf_u(lo) | ((unsigned)f2bf_u(hi) << 16);
#endif
}

__device__ __forceinline__ bf16x8 pack8(f32x4 a, f32x4 b) {
  u32x4 r;
  r[0] = pkbf2(a[0], a[1]);
  r[1] = pkbf2(a[2], a[3]);
  r[2] = pkbf2(b[0], b[1]);
  r[3] = pkbf2(b[2], b[3]);
  return __builtin_bit_cast(bf16x8, r);
}

// ---------------------------------------------------------------------------
// WMMA fragment loads from LDS (CDNA5 wave32 layouts, 16-bit A/B, 16x16x32)
//   A (16x32, MxK): lane m = lane&15; halves 0..7 = k0..k0+7, halves 8..15 =
//                   k0+16..k0+23, k0 = (lane>>4)*8
//   B (32x16, KxN) stored LDS-transposed as [N][K]: lane n = lane&15;
//                   halves 0..7 = k0..k0+7, halves 8..15 = k0+16..k0+23, k0 = (lane>>4)*8
// ---------------------------------------------------------------------------
__device__ __forceinline__ v16bf load_frag_a(const __bf16* base, int row0, int lane) {
  int m  = row0 + (lane & 15);
  int k0 = (lane >> 4) << 3;                 // 0 or 8
  const __bf16* p = base + (size_t)m * BKP + k0;
  bf16x8 lo = *(const bf16x8*)(p);
  bf16x8 hi = *(const bf16x8*)(p + 16);
  v16bf r;
#pragma unroll
  for (int i = 0; i < 8; ++i) { r[i] = lo[i]; r[i + 8] = hi[i]; }
  return r;
}

__device__ __forceinline__ v16bf load_frag_b(const __bf16* base, int col0, int lane) {
  int n  = col0 + (lane & 15);
  int k0 = (lane >> 4) << 3;                 // 0 or 8
  const __bf16* p = base + (size_t)n * BKP + k0;
  bf16x8 lo = *(const bf16x8*)(p);
  bf16x8 hi = *(const bf16x8*)(p + 16);
  v16bf r;
#pragma unroll
  for (int i = 0; i < 8; ++i) { r[i] = lo[i]; r[i + 8] = hi[i]; }
  return r;
}

// ---------------------------------------------------------------------------
// Generic bf16-WMMA GEMM with fp32 accumulate.
//   C[z] = alpha * A[z] @ B[z] (+ bias) (ReLU)
//
// Guard-free staging contract (holds for every launch in this model):
//   * K is a multiple of 32; M a multiple of 128 (1024 / 4096).
//   * N is a multiple of 8 (64 / 512 / 1024 / 2048 / 25000): every 8-wide
//     staging segment is fully in-range or fully out-of-range; fully OOB
//     segments are clamped to a safe address and deposit garbage only in LDS
//     columns whose outputs the epilogue masks (WMMA outputs are
//     column-independent -> no contamination).
//   * All row strides and tile offsets are 16-byte aligned -> b128 loads.
// ---------------------------------------------------------------------------
template <bool TRANSB, bool BIAS, bool RELU>
__global__ __launch_bounds__(256)
void gemm_bf16_wmma_kernel(const float* __restrict__ A, const float* __restrict__ Bm,
                           const float* __restrict__ bias, float* __restrict__ C,
                           int M, int N, int K, int lda, int ldb, int ldc,
                           int bH,
                           long long sAb, long long sAh,
                           long long sBb, long long sBh,
                           long long sCb, long long sCh,
                           float alpha) {
  __shared__ __align__(16) __bf16 lsA[BM][BKP];
  __shared__ __align__(16) __bf16 lsB[BN][BKP];

  const int z  = blockIdx.z;
  const int zb = z / bH, zh = z % bH;
  A  += (size_t)zb * sAb + (size_t)zh * sAh;
  Bm += (size_t)zb * sBb + (size_t)zh * sBh;
  C  += (size_t)zb * sCb + (size_t)zh * sCh;

  const int tid  = threadIdx.x;
  const int lane = tid & 31;
  const int wid  = tid >> 5;        // 0..7
  const int wm   = (wid & 1) * 64;  // wave row block: 0 or 64 (4 m-tiles)
  const int wn   = (wid >> 1) * 32; // wave col block: 0..96    (2 n-tiles)

  const int m0 = blockIdx.y * BM;
  const int n0 = blockIdx.x * BN;

  // ---- loop-invariant staging addresses (branch-free, clamped) ----
  // A tile: 128 rows x 32 k; thread: row = tid>>1, k segment = (tid&1)*16
  const int ar  = tid >> 1;
  const int aks = (tid & 1) * 16;
  int agm = m0 + ar; agm = agm < M ? agm : M - 1;
  const float* aBase = A + (size_t)agm * lda + aks;

  // B tile geometry
  const float* bBase;
  int bks = 0, bns = 0, bkk = 0;
  if (TRANSB) {              // B given [N][K]: same pattern as A
    int r = tid >> 1;
    bks = (tid & 1) * 16;
    int gn = n0 + r; gn = gn < N ? gn : N - 1;
    bBase = Bm + (size_t)gn * ldb + bks;
    bns = r;
  } else {                   // B given [K][N]: 2(k) x 8(n) block per thread
    bkk = (tid & 15) * 2;          // k pair 0..30
    bns = (tid >> 4) * 8;          // n segment 0..120
    int nb = n0 + bns;
    if (nb + 8 > N) nb = N - 8;    // fully-OOB segment clamp (N % 8 == 0)
    bBase = Bm + (size_t)bkk * ldb + nb;
  }

  const v8f vzero = {0.f, 0.f, 0.f, 0.f, 0.f, 0.f, 0.f, 0.f};
  v8f acc[4][2];
#pragma unroll
  for (int tm = 0; tm < 4; ++tm)
#pragma unroll
    for (int tn = 0; tn < 2; ++tn) acc[tm][tn] = vzero;

  for (int k0 = 0; k0 < K; k0 += BKt) {
    __syncthreads();
    // --- stage A tile: 4x global b128 -> 2x ds b128 ---
    {
      const f32x4* ap = (const f32x4*)(aBase + k0);
      f32x4 f0 = ap[0], f1 = ap[1], f2 = ap[2], f3 = ap[3];
      if (k0 + BKt < K) __builtin_prefetch(aBase + k0 + BKt, 0, 3);
      *(bf16x8*)&lsA[ar][aks]     = pack8(f0, f1);
      *(bf16x8*)&lsA[ar][aks + 8] = pack8(f2, f3);
    }
    // --- stage B tile into [N][K] layout ---
    if (TRANSB) {
      const f32x4* bp = (const f32x4*)(bBase + k0);
      f32x4 f0 = bp[0], f1 = bp[1], f2 = bp[2], f3 = bp[3];
      if (k0 + BKt < K) __builtin_prefetch(bBase + k0 + BKt, 0, 3);
      *(bf16x8*)&lsB[bns][bks]     = pack8(f0, f1);
      *(bf16x8*)&lsB[bns][bks + 8] = pack8(f2, f3);
    } else {
      const float* brow = bBase + (size_t)k0 * ldb;
      const f32x4* b0 = (const f32x4*)(brow);
      const f32x4* b1 = (const f32x4*)(brow + ldb);
      f32x4 r00 = b0[0], r01 = b0[1], r10 = b1[0], r11 = b1[1];
      if (k0 + BKt < K) __builtin_prefetch(brow + (size_t)BKt * ldb, 0, 3);
#pragma unroll
      for (int j = 0; j < 8; ++j) {
        float vlo = (j < 4) ? r00[j] : r01[j - 4];   // k   = bkk
        float vhi = (j < 4) ? r10[j] : r11[j - 4];   // k+1 = bkk+1
        *(unsigned*)&lsB[bns + j][bkk] = pkbf2(vlo, vhi);  // (k,k+1) contiguous
      }
    }
    __syncthreads();

    // --- fragments + 8 WMMAs per wave (wave tile 64x32) ---
    v16bf af[4], bfv[2];
#pragma unroll
    for (int tm = 0; tm < 4; ++tm) af[tm] = load_frag_a(&lsA[0][0], wm + tm * 16, lane);
#pragma unroll
    for (int tn = 0; tn < 2; ++tn) bfv[tn] = load_frag_b(&lsB[0][0], wn + tn * 16, lane);
#pragma unroll
    for (int tm = 0; tm < 4; ++tm)
#pragma unroll
      for (int tn = 0; tn < 2; ++tn)
        acc[tm][tn] = __builtin_amdgcn_wmma_f32_16x16x32_bf16(
            false, af[tm], false, bfv[tn], (short)0, acc[tm][tn], false, false);
  }

  // --- epilogue: C/D layout: VGPR r -> M = r + (lane>=16 ? 8:0), N = lane&15 ---
  const int cn    = lane & 15;
  const int rhalf = (lane >> 4) << 3;
#pragma unroll
  for (int tm = 0; tm < 4; ++tm) {
#pragma unroll
    for (int tn = 0; tn < 2; ++tn) {
#pragma unroll
      for (int r = 0; r < 8; ++r) {
        int rowg = m0 + wm + tm * 16 + r + rhalf;
        int colg = n0 + wn + tn * 16 + cn;
        if (rowg < M && colg < N) {
          float v = acc[tm][tn][r] * alpha;
          if (BIAS) v += bias[colg];
          if (RELU) v = fmaxf(v, 0.f);
          C[(size_t)rowg * ldc + colg] = v;
        }
      }
    }
  }
}

// ---------------------------------------------------------------------------
// Embedding + positional encoding: x[b,s,:] = emb[tok[b,s],:] + pe[s,:]
// ---------------------------------------------------------------------------
__global__ __launch_bounds__(256)
void embed_pe_kernel(const int* __restrict__ tok, const float* __restrict__ emb,
                     const float* __restrict__ pe, float* __restrict__ x) {
  size_t i  = (size_t)blockIdx.x * 256 + threadIdx.x;
  int d     = (int)(i % TD);
  size_t bs = i / TD;
  int s     = (int)(bs % TS);
  int t     = tok[bs];
  x[i] = emb[(size_t)t * TD + d] + pe[(size_t)s * TD + d];
}

// ---------------------------------------------------------------------------
// Fused key-padding + causal mask + softmax over rows of P [B*H, Sq, Sk]
// ---------------------------------------------------------------------------
__global__ __launch_bounds__(256)
void softmax_mask_kernel(float* __restrict__ P, const int* __restrict__ keyTok,
                         int Sq, int Sk, int H, int causal) {
  __shared__ float red[256];
  const int tid = threadIdx.x;
  const int bid = blockIdx.x;
  const int q   = bid % Sq;
  const int bh  = bid / Sq;
  const int b   = bh / H;
  float* row = P + (size_t)bh * Sq * Sk + (size_t)q * Sk;
  const int* kt = keyTok + (size_t)b * Sk;

  float vals[4];
  bool  ok[4];
  float mx = -3.4e38f;
#pragma unroll
  for (int j = 0; j < 4; ++j) {
    int k = tid + j * 256;
    bool v = (k < Sk) && (kt[k] != PAD_TOK) && (!causal || k <= q);
    ok[j]   = v;
    vals[j] = v ? row[k] : -3.4e38f;
    mx = fmaxf(mx, vals[j]);
  }
  red[tid] = mx; __syncthreads();
  for (int s = 128; s > 0; s >>= 1) {
    if (tid < s) red[tid] = fmaxf(red[tid], red[tid + s]);
    __syncthreads();
  }
  float m = red[0]; __syncthreads();

  float sum = 0.f;
#pragma unroll
  for (int j = 0; j < 4; ++j) {
    float e = ok[j] ? __expf(vals[j] - m) : 0.f;
    vals[j] = e; sum += e;
  }
  red[tid] = sum; __syncthreads();
  for (int s = 128; s > 0; s >>= 1) {
    if (tid < s) red[tid] += red[tid + s];
    __syncthreads();
  }
  float inv = red[0] > 0.f ? 1.f / red[0] : 0.f;
#pragma unroll
  for (int j = 0; j < 4; ++j) {
    int k = tid + j * 256;
    if (k < Sk) row[k] = vals[j] * inv;
  }
}

// ---------------------------------------------------------------------------
// out = LayerNorm(x + a) * g + b ; one block (256 thr) per row of D=512
// ---------------------------------------------------------------------------
__global__ __launch_bounds__(256)
void add_ln_kernel(const float* __restrict__ x, const float* __restrict__ a,
                   const float* __restrict__ g, const float* __restrict__ bta,
                   float* __restrict__ out) {
  __shared__ float red[256];
  const int tid = threadIdx.x;
  const size_t row = blockIdx.x;
  const float* xr = x + row * TD;
  const float* ar = a + row * TD;
  float v0 = xr[tid] + ar[tid];
  float v1 = xr[tid + 256] + ar[tid + 256];

  red[tid] = v0 + v1; __syncthreads();
  for (int s = 128; s > 0; s >>= 1) {
    if (tid < s) red[tid] += red[tid + s];
    __syncthreads();
  }
  float mean = red[0] * (1.f / TD); __syncthreads();

  float d0 = v0 - mean, d1 = v1 - mean;
  red[tid] = d0 * d0 + d1 * d1; __syncthreads();
  for (int s = 128; s > 0; s >>= 1) {
    if (tid < s) red[tid] += red[tid + s];
    __syncthreads();
  }
  float rstd = rsqrtf(red[0] * (1.f / TD) + 1e-5f);
  out[row * TD + tid]       = d0 * rstd * g[tid] + bta[tid];
  out[row * TD + tid + 256] = d1 * rstd * g[tid + 256] + bta[tid + 256];
}

// ---------------------------------------------------------------------------
// Host-side GEMM dispatch
// ---------------------------------------------------------------------------
static void launch_gemm(hipStream_t st, const float* A, const float* B,
                        const float* bias, float* C,
                        int M, int N, int K, int lda, int ldb, int ldc,
                        int batches, int bH,
                        long long sAb, long long sAh,
                        long long sBb, long long sBh,
                        long long sCb, long long sCh,
                        float alpha, bool transB, bool useBias, bool relu) {
  dim3 grid((N + BN - 1) / BN, (M + BM - 1) / BM, batches);
  dim3 blk(256);
  if (transB) {
    gemm_bf16_wmma_kernel<true, false, false><<<grid, blk, 0, st>>>(
        A, B, bias, C, M, N, K, lda, ldb, ldc, bH, sAb, sAh, sBb, sBh, sCb, sCh, alpha);
  } else if (useBias && relu) {
    gemm_bf16_wmma_kernel<false, true, true><<<grid, blk, 0, st>>>(
        A, B, bias, C, M, N, K, lda, ldb, ldc, bH, sAb, sAh, sBb, sBh, sCb, sCh, alpha);
  } else if (useBias) {
    gemm_bf16_wmma_kernel<false, true, false><<<grid, blk, 0, st>>>(
        A, B, bias, C, M, N, K, lda, ldb, ldc, bH, sAb, sAh, sBb, sBh, sCb, sCh, alpha);
  } else {
    gemm_bf16_wmma_kernel<false, false, false><<<grid, blk, 0, st>>>(
        A, B, bias, C, M, N, K, lda, ldb, ldc, bH, sAb, sAh, sBb, sBh, sCb, sCh, alpha);
  }
}

// Multi-head attention: out = proj( softmax(mask(QK^T/sqrt(dk))) @ V )
static void run_mha(hipStream_t st, const float* xq, const float* xkv,
                    const float* W, const float* bvec, const int* keyTok, bool causal,
                    float* Qb, float* Kb, float* Vb, float* P, float* Ob, float* out) {
  const int M = TB * TS;
  const long long DD = (long long)TD * TD;
  launch_gemm(st, xq,  W + 0 * DD, bvec + 0 * TD, Qb, M, TD, TD, TD, TD, TD,
              1, 1, 0, 0, 0, 0, 0, 0, 1.f, false, true, false);
  launch_gemm(st, xkv, W + 1 * DD, bvec + 1 * TD, Kb, M, TD, TD, TD, TD, TD,
              1, 1, 0, 0, 0, 0, 0, 0, 1.f, false, true, false);
  launch_gemm(st, xkv, W + 2 * DD, bvec + 2 * TD, Vb, M, TD, TD, TD, TD, TD,
              1, 1, 0, 0, 0, 0, 0, 0, 1.f, false, true, false);
  // scores = Q @ K^T / 8, batched over B*H (views into [B,S,H*DK])
  launch_gemm(st, Qb, Kb, nullptr, P, TS, TS, TDK, TD, TD, TS,
              TB * THd, THd,
              (long long)TS * TD, TDK,
              (long long)TS * TD, TDK,
              (long long)THd * TS * TS, (long long)TS * TS,
              0.125f, true, false, false);
  softmax_mask_kernel<<<dim3(TB * THd * TS), 256, 0, st>>>(P, keyTok, TS, TS, THd,
                                                           causal ? 1 : 0);
  // O = P @ V
  launch_gemm(st, P, Vb, nullptr, Ob, TS, TDK, TS, TS, TD, TD,
              TB * THd, THd,
              (long long)THd * TS * TS, (long long)TS * TS,
              (long long)TS * TD, TDK,
              (long long)TS * TD, TDK,
              1.f, false, false, false);
  launch_gemm(st, Ob, W + 3 * DD, bvec + 3 * TD, out, M, TD, TD, TD, TD, TD,
              1, 1, 0, 0, 0, 0, 0, 0, 1.f, false, true, false);
}

// ---------------------------------------------------------------------------
// Entry point
// ---------------------------------------------------------------------------
extern "C" void kernel_launch(void* const* d_in, const int* in_sizes, int n_in,
                              void* d_out, int out_size, void* d_ws, size_t ws_size,
                              hipStream_t stream) {
  (void)in_sizes; (void)n_in; (void)out_size; (void)ws_size;

  const int*   src       = (const int*)  d_in[0];
  const int*   trg       = (const int*)  d_in[1];
  const float* emb_src   = (const float*)d_in[2];
  const float* emb_trg   = (const float*)d_in[3];
  const float* pe        = (const float*)d_in[4];
  const float* enc_attn_W = (const float*)d_in[5];
  const float* enc_attn_b = (const float*)d_in[6];
  const float* enc_ln_g   = (const float*)d_in[7];
  const float* enc_ln_b   = (const float*)d_in[8];
  const float* enc_ff_W1  = (const float*)d_in[9];
  const float* enc_ff_b1  = (const float*)d_in[10];
  const float* enc_ff_W2  = (const float*)d_in[11];
  const float* enc_ff_b2  = (const float*)d_in[12];
  const float* dec_self_W = (const float*)d_in[13];
  const float* dec_self_b = (const float*)d_in[14];
  const float* dec_cross_W = (const float*)d_in[15];
  const float* dec_cross_b = (const float*)d_in[16];
  const float* dec_ln_g   = (const float*)d_in[17];
  const float* dec_ln_b   = (const float*)d_in[18];
  const float* dec_ff_W1  = (const float*)d_in[19];
  const float* dec_ff_b1  = (const float*)d_in[20];
  const float* dec_ff_W2  = (const float*)d_in[21];
  const float* dec_ff_b2  = (const float*)d_in[22];
  const float* fc_W       = (const float*)d_in[23];
  const float* fc_b       = (const float*)d_in[24];
  float* out = (float*)d_out;

  // workspace carve-up (floats)
  const size_t BSD = (size_t)TB * TS * TD;   // 2M
  const size_t BSF = (size_t)TB * TS * TF;   // 8M
  float* ws  = (float*)d_ws;
  float* X   = ws;              // encoder activations
  float* Y   = X + BSD;         // decoder activations
  float* Qb  = Y + BSD;
  float* Kb  = Qb + BSD;
  float* Vb  = Kb + BSD;
  float* Ob  = Vb + BSD;        // concat attention output
  float* A2  = Ob + BSD;        // attn-proj / ffn output
  float* HID = A2 + BSD;        // ffn hidden [B*S, F]
  float* P   = HID + BSF;       // attention scores [B*H, S, S]

  const int M = TB * TS;
  const dim3 embGrid((unsigned)(BSD / 256));

  // ----- encoder -----
  embed_pe_kernel<<<embGrid, 256, 0, stream>>>(src, emb_src, pe, X);
  for (int i = 0; i < TL; ++i) {
    const float* Wl = enc_attn_W + (size_t)i * 4 * TD * TD;
    const float* bl = enc_attn_b + (size_t)i * 4 * TD;
    run_mha(stream, X, X, Wl, bl, src, /*causal=*/false, Qb, Kb, Vb, P, Ob, A2);
    add_ln_kernel<<<M, 256, 0, stream>>>(X, A2,
        enc_ln_g + ((size_t)i * 2 + 0) * TD, enc_ln_b + ((size_t)i * 2 + 0) * TD, X);
    launch_gemm(stream, X, enc_ff_W1 + (size_t)i * TD * TF, enc_ff_b1 + (size_t)i * TF,
                HID, M, TF, TD, TD, TF, TF, 1, 1, 0, 0, 0, 0, 0, 0,
                1.f, false, true, true);
    launch_gemm(stream, HID, enc_ff_W2 + (size_t)i * TF * TD, enc_ff_b2 + (size_t)i * TD,
                A2, M, TD, TF, TF, TD, TD, 1, 1, 0, 0, 0, 0, 0, 0,
                1.f, false, true, false);
    add_ln_kernel<<<M, 256, 0, stream>>>(X, A2,
        enc_ln_g + ((size_t)i * 2 + 1) * TD, enc_ln_b + ((size_t)i * 2 + 1) * TD, X);
  }

  // ----- decoder -----
  embed_pe_kernel<<<embGrid, 256, 0, stream>>>(trg, emb_trg, pe, Y);
  for (int i = 0; i < TL; ++i) {
    const float* Wsf = dec_self_W  + (size_t)i * 4 * TD * TD;
    const float* bsf = dec_self_b  + (size_t)i * 4 * TD;
    const float* Wcr = dec_cross_W + (size_t)i * 4 * TD * TD;
    const float* bcr = dec_cross_b + (size_t)i * 4 * TD;
    run_mha(stream, Y, Y, Wsf, bsf, trg, /*causal=*/true, Qb, Kb, Vb, P, Ob, A2);
    add_ln_kernel<<<M, 256, 0, stream>>>(Y, A2,
        dec_ln_g + ((size_t)i * 3 + 0) * TD, dec_ln_b + ((size_t)i * 3 + 0) * TD, Y);
    run_mha(stream, Y, X, Wcr, bcr, src, /*causal=*/false, Qb, Kb, Vb, P, Ob, A2);
    add_ln_kernel<<<M, 256, 0, stream>>>(Y, A2,
        dec_ln_g + ((size_t)i * 3 + 1) * TD, dec_ln_b + ((size_t)i * 3 + 1) * TD, Y);
    launch_gemm(stream, Y, dec_ff_W1 + (size_t)i * TD * TF, dec_ff_b1 + (size_t)i * TF,
                HID, M, TF, TD, TD, TF, TF, 1, 1, 0, 0, 0, 0, 0, 0,
                1.f, false, true, true);
    launch_gemm(stream, HID, dec_ff_W2 + (size_t)i * TF * TD, dec_ff_b2 + (size_t)i * TD,
                A2, M, TD, TF, TF, TD, TD, 1, 1, 0, 0, 0, 0, 0, 0,
                1.f, false, true, false);
    add_ln_kernel<<<M, 256, 0, stream>>>(Y, A2,
        dec_ln_g + ((size_t)i * 3 + 2) * TD, dec_ln_b + ((size_t)i * 3 + 2) * TD, Y);
  }

  // ----- final vocab projection: [B*S, D] @ [D, V] + b -----
  launch_gemm(stream, Y, fc_W, fc_b, out, M, TV, TD, TD, TV, TV,
              1, 1, 0, 0, 0, 0, 0, 0, 1.f, false, true, false);
}
